// LSRE_CANN_60455959658875
// MI455X (gfx1250) — compile-verified, matrix operand order checked by reading, not run on hardware
//
#include <hip/hip_runtime.h>
#include <hip/hip_bf16.h>
#include <math.h>

typedef __attribute__((ext_vector_type(16))) _Float16 v16h;
typedef __attribute__((ext_vector_type(8)))  float    v8f;

#define NLAT   16
#define LDIM   256
#define DEPTH  4
#define NA     128
#define FLATD  4096
#define SHD    32

// ---------------------------------------------------------------------------
// A/B fragment loader for V_WMMA_F32_16X16X32_F16.
// 16-bit A 16x32 layout (ISA 7.12.2): lane L holds row M=L&15;
//   elems 0..7  = K = kk + (L>>4)*8 + j
//   elems 8..15 = K = kk + 16 + (L>>4)*8 + j
// B is loaded "column-major": lane L holds W row (nTile*16 + (L&15)), same K
// pattern, so row-major weights feed B directly.
// ---------------------------------------------------------------------------
__device__ __forceinline__ v16h frag16(const float* rowp, int kk, int half) {
  v16h r;
  const float* p = rowp + kk + half * 8;
#pragma unroll
  for (int j = 0; j < 8; ++j) {
    r[j]     = (_Float16)p[j];
    r[j + 8] = (_Float16)p[j + 16];
  }
  return r;
}

__device__ __forceinline__ v8f wmma_f16(v16h a, v16h b, v8f c) {
  return __builtin_amdgcn_wmma_f32_16x16x32_f16(false, a, false, b,
                                                (short)0, c, false, false);
}

// out(16x256) = in(16x256) @ W(256x256)^T [+bias][+resid][relu]
// one wave computes N-tiles {2w, 2w+1}; stride of in/W/out rows = 256.
__device__ __forceinline__ void mm16(const float* in, const float* W,
                                     const float* bias, const float* resid,
                                     float* out, bool relu, int wave, int lane) {
  const int half = lane >> 4, l16 = lane & 15;
  const int n0 = wave * 2, n1 = n0 + 1;
  v8f acc0 = {}, acc1 = {};
  const float* arow  = in + l16 * LDIM;
  const float* brow0 = W + (n0 * 16 + l16) * LDIM;
  const float* brow1 = W + (n1 * 16 + l16) * LDIM;
#pragma unroll
  for (int kk = 0; kk < LDIM; kk += 32) {
    v16h a  = frag16(arow, kk, half);
    v16h b0 = frag16(brow0, kk, half);
    v16h b1 = frag16(brow1, kk, half);
    acc0 = wmma_f16(a, b0, acc0);
    acc1 = wmma_f16(a, b1, acc1);
  }
#pragma unroll
  for (int v = 0; v < 8; ++v) {
    int m  = v + half * 8;
    int c0 = n0 * 16 + l16, c1 = n1 * 16 + l16;
    float x0 = acc0[v], x1 = acc1[v];
    if (bias)  { x0 += bias[c0];            x1 += bias[c1]; }
    if (resid) { x0 += resid[m * LDIM + c0]; x1 += resid[m * LDIM + c1]; }
    if (relu)  { x0 = fmaxf(x0, 0.f);       x1 = fmaxf(x1, 0.f); }
    out[m * LDIM + c0] = x0;
    out[m * LDIM + c1] = x1;
  }
}

__device__ __forceinline__ void layernorm16(const float* u, float* out,
                                            const float* g, const float* b,
                                            int wave, int lane) {
#pragma unroll
  for (int rr = 0; rr < 2; ++rr) {
    int r = wave * 2 + rr;
    float s = 0.f, s2 = 0.f;
#pragma unroll
    for (int c = lane; c < LDIM; c += 32) {
      float x = u[r * LDIM + c];
      s += x; s2 += x * x;
    }
#pragma unroll
    for (int mask = 1; mask < 32; mask <<= 1) {
      s  += __shfl_xor(s, mask, 32);
      s2 += __shfl_xor(s2, mask, 32);
    }
    float mean = s * (1.f / LDIM);
    float var  = s2 * (1.f / LDIM) - mean * mean;
    float inv  = rsqrtf(var + 1e-5f);
#pragma unroll
    for (int c = lane; c < LDIM; c += 32)
      out[r * LDIM + c] = (u[r * LDIM + c] - mean) * inv * g[c] + b[c];
  }
}

// ---------------------------------------------------------------------------
// Kernel 1: DEPTH latent self-attn blocks. 1 workgroup (8 waves) per asset.
// ---------------------------------------------------------------------------
__global__ __launch_bounds__(256) void latent_blocks_kernel(
    const float* __restrict__ z,
    const float* __restrict__ s_wq, const float* __restrict__ s_wk,
    const float* __restrict__ s_wv, const float* __restrict__ s_wo,
    const float* __restrict__ s_bo,
    const float* __restrict__ s_ln1_g, const float* __restrict__ s_ln1_b,
    const float* __restrict__ s_ff_w1, const float* __restrict__ s_ff_b1,
    const float* __restrict__ s_ff_w2, const float* __restrict__ s_ff_b2,
    const float* __restrict__ s_ln2_g, const float* __restrict__ s_ln2_b,
    float* __restrict__ zf) {
  __shared__ float zS[NLAT * LDIM];
  __shared__ float qS[NLAT * LDIM];
  __shared__ float kS[NLAT * LDIM];
  __shared__ float vS[NLAT * LDIM];
  __shared__ float tS[NLAT * LDIM];
  __shared__ float pS[8 * NLAT * NLAT];

  const int tid  = threadIdx.x;
  const int wave = tid >> 5;
  const int lane = tid & 31;
  const int half = lane >> 4;
  const int l16  = lane & 15;
  const int asset = blockIdx.x;

  for (int i = tid; i < NLAT * LDIM; i += 256)
    zS[i] = z[asset * NLAT * LDIM + i];
  __syncthreads();

  for (int d = 0; d < DEPTH; ++d) {
    const float* wq  = s_wq + d * LDIM * LDIM;
    const float* wk  = s_wk + d * LDIM * LDIM;
    const float* wv  = s_wv + d * LDIM * LDIM;
    const float* wo  = s_wo + d * LDIM * LDIM;
    const float* bo  = s_bo + d * LDIM;
    const float* g1  = s_ln1_g + d * LDIM;
    const float* b1  = s_ln1_b + d * LDIM;
    const float* fw1 = s_ff_w1 + d * LDIM * LDIM;
    const float* fb1 = s_ff_b1 + d * LDIM;
    const float* fw2 = s_ff_w2 + d * LDIM * LDIM;
    const float* fb2 = s_ff_b2 + d * LDIM;
    const float* g2  = s_ln2_g + d * LDIM;
    const float* b2  = s_ln2_b + d * LDIM;

    // Q, K, V projections (no bias in reference)
    mm16(zS, wq, nullptr, nullptr, qS, false, wave, lane);
    mm16(zS, wk, nullptr, nullptr, kS, false, wave, lane);
    mm16(zS, wv, nullptr, nullptr, vS, false, wave, lane);
    __syncthreads();

    // 8-head attention over 16 tokens; head = wave id.
    {
      const int h = wave;
      const float scale = 0.17677669529663687f;  // 32^-0.5
      float sv[8];
#pragma unroll
      for (int t = 0; t < 8; ++t) {
        int i = 2 * t + half;      // query row
        int j = l16;               // key row
        const float* qp = &qS[i * LDIM + h * SHD];
        const float* kp = &kS[j * LDIM + h * SHD];
        float s = 0.f;
#pragma unroll
        for (int c = 0; c < SHD; ++c) s += qp[c] * kp[c];
        sv[t] = s * scale;
      }
#pragma unroll
      for (int t = 0; t < 8; ++t) {
        float m = sv[t];
#pragma unroll
        for (int mask = 1; mask < 16; mask <<= 1)
          m = fmaxf(m, __shfl_xor(m, mask, 32));
        float e = __expf(sv[t] - m);
        float sum = e;
#pragma unroll
        for (int mask = 1; mask < 16; mask <<= 1)
          sum += __shfl_xor(sum, mask, 32);
        int i = 2 * t + half;
        pS[h * 256 + i * 16 + l16] = e / sum;
      }
      __syncthreads();
      // o[i][h*32+lane] = sum_j p[i][j] * v[j][h*32+lane]
#pragma unroll
      for (int i = 0; i < NLAT; ++i) {
        const float* pp = &pS[h * 256 + i * 16];
        float acc = 0.f;
#pragma unroll
        for (int j = 0; j < NLAT; ++j)
          acc += pp[j] * vS[j * LDIM + h * SHD + lane];
        tS[i * LDIM + h * SHD + lane] = acc;
      }
    }
    __syncthreads();

    // O projection + bias + residual -> qS ; LN1 -> zS
    mm16(tS, wo, bo, zS, qS, false, wave, lane);
    __syncthreads();
    layernorm16(qS, zS, g1, b1, wave, lane);
    __syncthreads();

    // FF: relu(z@fw1^T+fb1) -> kS ; kS@fw2^T+fb2 + z -> vS ; LN2 -> zS
    mm16(zS, fw1, fb1, nullptr, kS, true, wave, lane);
    __syncthreads();
    mm16(kS, fw2, fb2, zS, vS, false, wave, lane);
    __syncthreads();
    layernorm16(vS, zS, g2, b2, wave, lane);
    __syncthreads();
  }

  for (int i = tid; i < NLAT * LDIM; i += 256)
    zf[asset * NLAT * LDIM + i] = zS[i];
}

// ---------------------------------------------------------------------------
// Kernel 2: Q/K/V = zf(128x4096) @ W(4096x4096)^T + b.
// grid=(256 n-tiles, 3 matrices); 8 waves = 8 M-tiles (all 128 rows),
// so every weight element is read from HBM exactly once.
// ---------------------------------------------------------------------------
__global__ __launch_bounds__(256) void qkv_kernel(
    const float* __restrict__ zf,
    const float* __restrict__ qw, const float* __restrict__ qb,
    const float* __restrict__ kw, const float* __restrict__ kb,
    const float* __restrict__ vw, const float* __restrict__ vb,
    float* __restrict__ Q, float* __restrict__ K, float* __restrict__ V) {
  const int nt = blockIdx.x;
  const int which = blockIdx.y;
  const float* W = (which == 0) ? qw : (which == 1) ? kw : vw;
  const float* B = (which == 0) ? qb : (which == 1) ? kb : vb;
  float*       O = (which == 0) ? Q : (which == 1) ? K : V;
  const int wave = threadIdx.x >> 5, lane = threadIdx.x & 31;
  const int half = lane >> 4, l16 = lane & 15;
  v8f acc = {};
  const float* arow = zf + (wave * 16 + l16) * FLATD;
  const float* brow = W + (nt * 16 + l16) * FLATD;
  for (int kk = 0; kk < FLATD; kk += 32) {
    v16h a = frag16(arow, kk, half);
    v16h b = frag16(brow, kk, half);
    acc = wmma_f16(a, b, acc);
  }
  const int col = nt * 16 + l16;
  const float bias = B[col];
#pragma unroll
  for (int v = 0; v < 8; ++v)
    O[(wave * 16 + v + half * 8) * FLATD + col] = acc[v] + bias;
}

// ---------------------------------------------------------------------------
// Kernel 3: P = softmax(Q @ K^T * 256^-0.5). One workgroup; each wave owns a
// 16-row stripe x all 128 columns (8 accumulators), fused softmax.
// ---------------------------------------------------------------------------
__global__ __launch_bounds__(256) void scores_kernel(
    const float* __restrict__ Q, const float* __restrict__ Km,
    float* __restrict__ P) {
  const int wave = threadIdx.x >> 5, lane = threadIdx.x & 31;
  const int half = lane >> 4, l16 = lane & 15;
  v8f acc[8];
#pragma unroll
  for (int t = 0; t < 8; ++t) acc[t] = (v8f){};
  const float* arow = Q + (wave * 16 + l16) * FLATD;
  for (int kk = 0; kk < FLATD; kk += 32) {
    v16h a = frag16(arow, kk, half);
#pragma unroll
    for (int t = 0; t < 8; ++t) {
      v16h b = frag16(Km + (t * 16 + l16) * FLATD, kk, half);
      acc[t] = wmma_f16(a, b, acc[t]);
    }
  }
  const float scale = 0.0625f;  // 256^-0.5
#pragma unroll
  for (int v = 0; v < 8; ++v) {
    float rmax = -3.4e38f;
#pragma unroll
    for (int t = 0; t < 8; ++t) rmax = fmaxf(rmax, acc[t][v] * scale);
#pragma unroll
    for (int mask = 1; mask < 16; mask <<= 1)
      rmax = fmaxf(rmax, __shfl_xor(rmax, mask, 32));
    float e[8], sum = 0.f;
#pragma unroll
    for (int t = 0; t < 8; ++t) {
      e[t] = __expf(acc[t][v] * scale - rmax);
      sum += e[t];
    }
#pragma unroll
    for (int mask = 1; mask < 16; mask <<= 1)
      sum += __shfl_xor(sum, mask, 32);
    const float inv = 1.f / sum;
    const int row = wave * 16 + v + half * 8;
#pragma unroll
    for (int t = 0; t < 8; ++t) P[row * 128 + t * 16 + l16] = e[t] * inv;
  }
}

// ---------------------------------------------------------------------------
// Kernel 4: H = P(128x128) @ V(128x4096). 256 blocks (n-tiles) x 8 waves
// (m-tiles); B frag gathered column-wise from V.
// ---------------------------------------------------------------------------
__global__ __launch_bounds__(256) void attnv_kernel(
    const float* __restrict__ P, const float* __restrict__ V,
    float* __restrict__ H) {
  const int nt = blockIdx.x;
  const int wave = threadIdx.x >> 5, lane = threadIdx.x & 31;
  const int half = lane >> 4, l16 = lane & 15;
  v8f acc = {};
  const float* arow = P + (wave * 16 + l16) * 128;
  const float* vcol = V + nt * 16 + l16;
#pragma unroll
  for (int kk = 0; kk < 128; kk += 32) {
    v16h a = frag16(arow, kk, half);
    v16h b;
#pragma unroll
    for (int j = 0; j < 8; ++j) {
      b[j]     = (_Float16)vcol[(kk + half * 8 + j) * FLATD];
      b[j + 8] = (_Float16)vcol[(kk + 16 + half * 8 + j) * FLATD];
    }
    acc = wmma_f16(a, b, acc);
  }
  const int col = nt * 16 + l16;
#pragma unroll
  for (int v = 0; v < 8; ++v)
    H[(wave * 16 + v + half * 8) * FLATD + col] = acc[v];
}

// ---------------------------------------------------------------------------
extern "C" void kernel_launch(void* const* d_in, const int* in_sizes, int n_in,
                              void* d_out, int out_size, void* d_ws,
                              size_t ws_size, hipStream_t stream) {
  (void)in_sizes; (void)n_in; (void)out_size; (void)ws_size;
  const float* z      = (const float*)d_in[1];
  const float* s_wq   = (const float*)d_in[15];
  const float* s_wk   = (const float*)d_in[16];
  const float* s_wv   = (const float*)d_in[17];
  const float* s_wo   = (const float*)d_in[18];
  const float* s_bo   = (const float*)d_in[19];
  const float* s_ln1g = (const float*)d_in[20];
  const float* s_ln1b = (const float*)d_in[21];
  const float* s_fw1  = (const float*)d_in[22];
  const float* s_fb1  = (const float*)d_in[23];
  const float* s_fw2  = (const float*)d_in[24];
  const float* s_fb2  = (const float*)d_in[25];
  const float* s_ln2g = (const float*)d_in[26];
  const float* s_ln2b = (const float*)d_in[27];
  const float* q_w    = (const float*)d_in[28];
  const float* q_b    = (const float*)d_in[29];
  const float* k_w    = (const float*)d_in[30];
  const float* k_b    = (const float*)d_in[31];
  const float* v_w    = (const float*)d_in[32];
  const float* v_b    = (const float*)d_in[33];

  float* out = (float*)d_out;
  float* ws  = (float*)d_ws;
  const size_t M = (size_t)NA * NLAT * LDIM;  // 524288
  float* zf = ws;
  float* Q  = ws + M;
  float* K  = ws + 2 * M;
  float* V  = ws + 3 * M;
  float* P  = ws + 4 * M;  // 128*128

  // z_prior: straight copy of z into second output slot.
  hipMemcpyAsync(out + M, z, M * sizeof(float), hipMemcpyDeviceToDevice,
                 stream);

  latent_blocks_kernel<<<NA, 256, 0, stream>>>(
      z, s_wq, s_wk, s_wv, s_wo, s_bo, s_ln1g, s_ln1b, s_fw1, s_fb1, s_fw2,
      s_fb2, s_ln2g, s_ln2b, zf);
  qkv_kernel<<<dim3(256, 3), 256, 0, stream>>>(zf, q_w, q_b, k_w, k_b, v_w,
                                               v_b, Q, K, V);
  scores_kernel<<<1, 256, 0, stream>>>(Q, K, P);
  attnv_kernel<<<256, 256, 0, stream>>>(P, V, out);
}
